// CEAlignment_14224931684631
// MI455X (gfx1250) — matrix operand order repaired and checked by reading, not compiled, not run on hardware
//
#include <hip/hip_runtime.h>
#include <hip/hip_bf16.h>

// CEAlignment pipeline for gfx1250 (MI455X), wave32 + WMMA.
// Workspace layout (needs ~24.5 MB of d_ws):
//   [0,4MB)    q1h  f16 [2048][16][64]  (standardized q1, scaled by 1/8)
//   [4,8MB)    q2h  f16 [2048][16][64]
//   [8,12MB)   hA   f32 [2048][512]  MLP ping
//   [12,16MB)  hB   f32 [2048][512]  MLP pong
//   [16,24MB)  yb   f32 [2048][1024] MLP output
//   [24MB+0)   v    f32 [16][2048]
//   [24MB+256K)u    f32 [16][2048]
//   [24MB+512K)mx   u32 [16] per-label max (monotone-encoded)

typedef _Float16 v16h __attribute__((ext_vector_type(16)));
typedef _Float16 v8h  __attribute__((ext_vector_type(8)));
typedef float    v8f  __attribute__((ext_vector_type(8)));

#define BATCH 2048
#define NLAB  16
#define EMBED 64

__device__ __forceinline__ v8f wmma_f16(v16h a, v16h b, v8f c) {
  // D = A(16x32 f16) x B(32x16 f16) + C(16x16 f32)
  return __builtin_amdgcn_wmma_f32_16x16x32_f16(false, a, false, b,
                                                (short)0, c, false, false);
}

// Monotone float<->uint map so atomicMax(u32) == float max.
__device__ __forceinline__ unsigned enc_f32(float f) {
  unsigned u = __float_as_uint(f);
  return (u & 0x80000000u) ? ~u : (u | 0x80000000u);
}
__device__ __forceinline__ float dec_f32(unsigned u) {
  return __uint_as_float((u & 0x80000000u) ? (u & 0x7FFFFFFFu) : ~u);
}

// Recompute one 16x16 affinity tile: rows from qA[btile], cols from qB[dtile],
// label c, K = EMBED = 64 via two v_wmma_f32_16x16x32_f16.
__device__ __forceinline__ v8f aff_tile(const _Float16* __restrict__ qA,
                                        const _Float16* __restrict__ qB,
                                        int btile, int dtile, int c, int lane) {
  v8f acc = {};
  const int g  = lane & 15;
  const int hi = lane >> 4;
  // A frag: lane holds row g; halves 0..15 -> K=hi*16+{0..15} (contiguous)
  const _Float16* abase = qA + ((size_t)(btile * 16 + g) * NLAB + c) * EMBED + hi * 16;
  // B frag: lane holds col g; halves 0..15 -> K=hi*16+{0..15} (contiguous!)
  const _Float16* bbase = qB + ((size_t)(dtile * 16 + g) * NLAB + c) * EMBED + hi * 16;
#pragma unroll
  for (int k0 = 0; k0 < EMBED; k0 += 32) {
    v8h alo = *(const v8h*)(abase + k0);
    v8h ahi = *(const v8h*)(abase + k0 + 8);
    v16h a;
    v16h b = *(const v16h*)(bbase + k0);
#pragma unroll
    for (int i = 0; i < 8; ++i) { a[i] = alo[i]; a[i + 8] = ahi[i]; }
    acc = wmma_f16(a, b, acc);
  }
  return acc;
}

// ---------------------------------------------------------------- MLP GEMM
// C[2048 x N] = act(X[2048 x K] * W[K x N] + bias). One wave computes a
// 64x16 output slab (4 M-tiles share each B fragment). f32->f16 on load.
__global__ __launch_bounds__(32) void k_gemm(const float* __restrict__ X,
                                             const float* __restrict__ W,
                                             const float* __restrict__ bias,
                                             float* __restrict__ C,
                                             int K, int N, int relu) {
  const int lane = threadIdx.x;
  const int g = lane & 15, hi = lane >> 4;
  const int col = blockIdx.x * 16 + g;
  const int rowbase = blockIdx.y * 64;
  v8f acc[4] = {};
  for (int k0 = 0; k0 < K; k0 += 32) {
    v16h b;
#pragma unroll
    for (int h = 0; h < 16; ++h)
      b[h] = (_Float16)W[(k0 + hi * 16 + h) * N + col];
#pragma unroll
    for (int t = 0; t < 4; ++t) {
      const float* ax = X + (size_t)(rowbase + t * 16 + g) * K + k0 + hi * 16;
      v16h a;
#pragma unroll
      for (int i = 0; i < 8; ++i) {
        a[i]     = (_Float16)ax[i];
        a[i + 8] = (_Float16)ax[8 + i];
      }
      acc[t] = wmma_f16(a, b, acc[t]);
    }
  }
  const float bv = bias[col];
#pragma unroll
  for (int t = 0; t < 4; ++t)
#pragma unroll
    for (int r = 0; r < 8; ++r) {
      const int row = rowbase + t * 16 + r + hi * 8;
      float v = acc[t][r] + bv;
      if (relu) v = fmaxf(v, 0.0f);
      C[(size_t)row * N + col] = v;
    }
}

// ------------------------------------------------- standardize + f16 pack
// One wave per (b,c): 64 embed values; unbiased var (ddof=1); scale folds
// the 1/sqrt(EMBED) of the affinity into q1.
__global__ __launch_bounds__(256) void k_std(const float* __restrict__ Y,
                                             _Float16* __restrict__ Q,
                                             float scale) {
  const int wglob = (int)(blockIdx.x * 8 + (threadIdx.x >> 5));
  const int lane = threadIdx.x & 31;
  const int b = wglob >> 4, c = wglob & 15;
  const float* y = Y + (size_t)b * (NLAB * EMBED) + c * EMBED;
  const float x0 = y[lane], x1 = y[lane + 32];
  float s = x0 + x1;
  s += __shfl_xor(s, 16, 32); s += __shfl_xor(s, 8, 32);
  s += __shfl_xor(s, 4, 32);  s += __shfl_xor(s, 2, 32);
  s += __shfl_xor(s, 1, 32);
  const float mu = s * (1.0f / 64.0f);
  const float d0 = x0 - mu, d1 = x1 - mu;
  float vs = d0 * d0 + d1 * d1;
  vs += __shfl_xor(vs, 16, 32); vs += __shfl_xor(vs, 8, 32);
  vs += __shfl_xor(vs, 4, 32);  vs += __shfl_xor(vs, 2, 32);
  vs += __shfl_xor(vs, 1, 32);
  const float var = vs * (1.0f / 63.0f);
  const float inv = scale * rsqrtf(var + 1e-8f);
  _Float16* q = Q + ((size_t)b * NLAB + c) * EMBED;
  q[lane]      = (_Float16)(d0 * inv);
  q[lane + 32] = (_Float16)(d1 * inv);
}

// ------------------------------------------------------- per-label max
__global__ __launch_bounds__(32) void k_max(const _Float16* __restrict__ q1,
                                            const _Float16* __restrict__ q2,
                                            unsigned* __restrict__ mx) {
  const int lane = threadIdx.x;
  v8f t = aff_tile(q1, q2, blockIdx.y, blockIdx.x, blockIdx.z, lane);
  float m = t[0];
#pragma unroll
  for (int r = 1; r < 8; ++r) m = fmaxf(m, t[r]);
  m = fmaxf(m, __shfl_xor(m, 16, 32));
  m = fmaxf(m, __shfl_xor(m, 8, 32));
  m = fmaxf(m, __shfl_xor(m, 4, 32));
  m = fmaxf(m, __shfl_xor(m, 2, 32));
  m = fmaxf(m, __shfl_xor(m, 1, 32));
  if (lane == 0) atomicMax(&mx[blockIdx.z], enc_f32(m));
}

// ------------------------------------------- A = exp(aff - max), stored f32
__global__ __launch_bounds__(32) void k_expA(const _Float16* __restrict__ q1,
                                             const _Float16* __restrict__ q2,
                                             const unsigned* __restrict__ mx,
                                             float* __restrict__ Aout) {
  const int lane = threadIdx.x;
  const int dtile = blockIdx.x, btile = blockIdx.y, c = blockIdx.z;
  v8f t = aff_tile(q1, q2, btile, dtile, c, lane);
  const float mC = dec_f32(mx[c]);
  const int col = dtile * 16 + (lane & 15);
  const int rb = btile * 16 + (lane >> 4) * 8;
#pragma unroll
  for (int r = 0; r < 8; ++r) {
    const size_t idx = ((size_t)(rb + r) * BATCH + col) * NLAB + c;
    Aout[idx] = __expf(t[r] - mC);
  }
}

// ------------------------------------------------- Sinkhorn matvec step
// vout[row] = marg[row]/(sum_col exp(aff-row,col - m)*vin[col] + 1e-10),
// with A-tiles recomputed by WMMA from L2-resident f16 q's.
// Transposed step: call with (qA,qB)=(q2h,q1h), (vin,vout)=(u,v).
__global__ __launch_bounds__(256) void k_sink(const _Float16* __restrict__ qA,
                                              const _Float16* __restrict__ qB,
                                              const unsigned* __restrict__ mx,
                                              const float* __restrict__ marg,
                                              const float* __restrict__ vin,
                                              float* __restrict__ vout) {
  __shared__ float red[8][16];
  const int c = blockIdx.y, btile = blockIdx.x;
  const int lane = threadIdx.x & 31, w = threadIdx.x >> 5;
  const float mC = dec_f32(mx[c]);
  const int g = lane & 15;
  float acc[8] = {0.f, 0.f, 0.f, 0.f, 0.f, 0.f, 0.f, 0.f};
  for (int dt = w; dt < BATCH / 16; dt += 8) {
    v8f t = aff_tile(qA, qB, btile, dt, c, lane);
    const float vcol = vin[c * BATCH + dt * 16 + g];
#pragma unroll
    for (int r = 0; r < 8; ++r) acc[r] += __expf(t[r] - mC) * vcol;
  }
#pragma unroll
  for (int r = 0; r < 8; ++r) {   // reduce over the 16 lanes sharing a row
    float s = acc[r];
    s += __shfl_xor(s, 1, 32);
    s += __shfl_xor(s, 2, 32);
    s += __shfl_xor(s, 4, 32);
    s += __shfl_xor(s, 8, 32);
    acc[r] = s;
  }
  if (g == 0) {
    const int hi = lane >> 4;
#pragma unroll
    for (int r = 0; r < 8; ++r) red[w][r + 8 * hi] = acc[r];
  }
  __syncthreads();
  if (threadIdx.x < 16) {
    float s = 0.f;
#pragma unroll
    for (int ww = 0; ww < 8; ++ww) s += red[ww][threadIdx.x];
    const int row = btile * 16 + (int)threadIdx.x;
    vout[c * BATCH + row] = marg[row * NLAB + c] / (s + 1e-10f);
  }
}

// ------------------------------------------------- P = u[:,None]*A*v[None,:]
__global__ __launch_bounds__(256) void k_P(const float* __restrict__ A,
                                           const float* __restrict__ u,
                                           const float* __restrict__ v,
                                           float* __restrict__ P) {
  const size_t idx = (size_t)blockIdx.x * 256 + threadIdx.x;
  const int c = (int)(idx & 15);
  const size_t bd = idx >> 4;
  const int d = (int)(bd & (BATCH - 1));
  const int b = (int)(bd >> 11);
  P[idx] = u[c * BATCH + b] * A[idx] * v[c * BATCH + d];
}

__global__ __launch_bounds__(256) void k_init(unsigned* __restrict__ mx,
                                              float* __restrict__ v) {
  const int i = (int)(blockIdx.x * 256 + threadIdx.x);
  if (i < 16) mx[i] = 0u;               // decodes below any finite float
  if (i < NLAB * BATCH) v[i] = 1.0f;    // Sinkhorn v0 = 1
}

extern "C" void kernel_launch(void* const* d_in, const int* in_sizes, int n_in,
                              void* d_out, int out_size, void* d_ws, size_t ws_size,
                              hipStream_t stream) {
  (void)in_sizes; (void)n_in; (void)out_size; (void)ws_size;
  const float* x1  = (const float*)d_in[0];
  const float* x2  = (const float*)d_in[1];
  const float* py1 = (const float*)d_in[2];
  const float* py2 = (const float*)d_in[3];

  char* ws = (char*)d_ws;
  _Float16* q1h = (_Float16*)(ws);
  _Float16* q2h = (_Float16*)(ws + (size_t)4  * 1024 * 1024);
  float*    hA  = (float*)   (ws + (size_t)8  * 1024 * 1024);
  float*    hB  = (float*)   (ws + (size_t)12 * 1024 * 1024);
  float*    yb  = (float*)   (ws + (size_t)16 * 1024 * 1024);
  float*    vv  = (float*)   (ws + (size_t)24 * 1024 * 1024);
  float*    uu  = (float*)   (ws + (size_t)24 * 1024 * 1024 + 256 * 1024);
  unsigned* mx  = (unsigned*)(ws + (size_t)24 * 1024 * 1024 + 512 * 1024);

  float* P = (float*)d_out;
  float* A = P + (size_t)BATCH * BATCH * NLAB;   // second output, 256 MB each

  k_init<<<128, 256, 0, stream>>>(mx, vv);

  // Two MLPs: 768 -> 512 -> 512 -> 512 (relu) -> 1024 (linear)
  for (int m = 0; m < 2; ++m) {
    const float* x = (m == 0) ? x1 : x2;
    const int base = 4 + m * 8;
    const float* W0 = (const float*)d_in[base + 0]; const float* b0 = (const float*)d_in[base + 1];
    const float* Wa = (const float*)d_in[base + 2]; const float* ba = (const float*)d_in[base + 3];
    const float* Wb = (const float*)d_in[base + 4]; const float* bb = (const float*)d_in[base + 5];
    const float* Wc = (const float*)d_in[base + 6]; const float* bc = (const float*)d_in[base + 7];
    k_gemm<<<dim3(32, 32), 32, 0, stream>>>(x,  W0, b0, hA, 768, 512, 1);
    k_gemm<<<dim3(32, 32), 32, 0, stream>>>(hA, Wa, ba, hB, 512, 512, 1);
    k_gemm<<<dim3(32, 32), 32, 0, stream>>>(hB, Wb, bb, hA, 512, 512, 1);
    k_gemm<<<dim3(64, 32), 32, 0, stream>>>(hA, Wc, bc, yb, 512, 1024, 0);
    // q1 carries the 1/sqrt(EMBED)=1/8 affinity scale
    k_std<<<4096, 256, 0, stream>>>(yb, (m == 0) ? q1h : q2h,
                                    (m == 0) ? 0.125f : 1.0f);
  }

  // Per-label global max of affinity, then materialize A = exp(aff - max)
  k_max <<<dim3(128, 128, 16), 32, 0, stream>>>(q1h, q2h, mx);
  k_expA<<<dim3(128, 128, 16), 32, 0, stream>>>(q1h, q2h, mx, A);

  // 20 Sinkhorn iterations (u then v), A recomputed from L2-resident q's
  for (int it = 0; it < 20; ++it) {
    k_sink<<<dim3(128, 16), 256, 0, stream>>>(q1h, q2h, mx, py1, vv, uu);
    k_sink<<<dim3(128, 16), 256, 0, stream>>>(q2h, q1h, mx, py2, uu, vv);
  }
  // final u = p/(A v + eps)
  k_sink<<<dim3(128, 16), 256, 0, stream>>>(q1h, q2h, mx, py1, vv, uu);

  k_P<<<(unsigned)((size_t)BATCH * BATCH * NLAB / 256), 256, 0, stream>>>(A, uu, vv, P);
}
